// Model_Comi_Rec_25769803776123
// MI455X (gfx1250) — matrix operand order repaired and checked by reading, not compiled, not run on hardware
//
#include <hip/hip_runtime.h>

typedef __attribute__((ext_vector_type(16))) _Float16 v16h;
typedef __attribute__((ext_vector_type(8)))  _Float16 v8h;
typedef __attribute__((ext_vector_type(4)))  _Float16 v4h;
typedef __attribute__((ext_vector_type(8)))  float    v8f;

#define B_TOT   1024
#define S_LEN   50
#define D_DIM   128
#define NCAP    4
#define KTOT    512   // NCAP * D_DIM

// ---------------------------------------------------------------------------
// Kernel 1: convert caps_w (f32, [s][k][d]) -> f16 copy in workspace
// ---------------------------------------------------------------------------
__global__ __launch_bounds__(256) void convert_w_kernel(
    const float* __restrict__ caps_w, _Float16* __restrict__ wf16, int n)
{
    int i4 = (blockIdx.x * 256 + threadIdx.x) * 4;
    if (i4 < n) {
        float4 v = *(const float4*)(caps_w + i4);
        v4h h;
        h.x = (_Float16)v.x; h.y = (_Float16)v.y;
        h.z = (_Float16)v.z; h.w = (_Float16)v.w;
        *(v4h*)(wf16 + i4) = h;
    }
}

// ---------------------------------------------------------------------------
// Kernel 2: hat[b, n, s, d] via WMMA f16->f32.
//   grid = (B/16, S); block = 256 threads (8 waves).
//   Each block: one 16-batch tile x one s; output 16 x 512 accumulated over
//   K = 128 with v_wmma_f32_16x16x32_f16.
//   wave w owns k-range [w*64, w*64+64) = 4 N-tiles of 16.
// ---------------------------------------------------------------------------
__global__ __launch_bounds__(256) void hat_wmma_kernel(
    const int*      __restrict__ hist_item,
    const float*    __restrict__ hist_mask,
    const float*    __restrict__ mid_emb,
    const _Float16* __restrict__ wf16,
    _Float16*       __restrict__ hatf)
{
    __shared__ _Float16 A_lds[16 * 136];   // 16 rows (batch) x 128 d, pad to 136

    const int btile = blockIdx.x;   // 0..63
    const int s     = blockIdx.y;   // 0..49
    const int tid   = threadIdx.x;

    // --- stage A = item_his (16 x 128) into LDS as f16 ---
    {
        const int bb = tid >> 4;            // 0..15 (batch within tile)
        const int d0 = (tid & 15) * 8;      // 0..120
        const int b  = btile * 16 + bb;
        const int row  = hist_item[b * S_LEN + s];
        const float msk = hist_mask[b * S_LEN + s];
        const float* src = mid_emb + (size_t)row * D_DIM + d0;
        float4 v0 = *(const float4*)(src);
        float4 v1 = *(const float4*)(src + 4);
        float vv[8] = {v0.x, v0.y, v0.z, v0.w, v1.x, v1.y, v1.z, v1.w};
        _Float16* dst = &A_lds[bb * 136 + d0];
        #pragma unroll
        for (int j = 0; j < 8; ++j) dst[j] = (_Float16)(vv[j] * msk);
    }
    __syncthreads();

    const int wave = tid >> 5;
    const int lane = tid & 31;
    const int m    = lane & 15;     // M row (A) / N col (B,C)
    const int grp  = lane >> 4;     // half-wave group
    const int kbase = wave * 64;

    v8f acc[4] = {v8f{}, v8f{}, v8f{}, v8f{}};
    const _Float16* wrow = wf16 + (size_t)s * KTOT * D_DIM;

    #pragma unroll
    for (int kk = 0; kk < D_DIM; kk += 32) {
        // A fragment (16-bit A 16x32 layout):
        //   lane<16 : M=lane,    halves 0-7 -> K=kk+0..7,  halves 8-15 -> K=kk+16..23
        //   lane>=16: M=lane-16, halves 0-7 -> K=kk+8..15, halves 8-15 -> K=kk+24..31
        const _Float16* ap = &A_lds[m * 136 + kk + grp * 8];
        v8h alo = *(const v8h*)(ap);
        v8h ahi = *(const v8h*)(ap + 16);
        v16h a = __builtin_shufflevector(alo, ahi,
                   0, 1, 2, 3, 4, 5, 6, 7, 8, 9, 10, 11, 12, 13, 14, 15);

        #pragma unroll
        for (int t = 0; t < 4; ++t) {
            // B fragment (32x16): lane<16: N=lane,K=kk+0..7,16..23 ; lane>=16: K=kk+8..15,24..31
            const int k = kbase + t * 16 + m;
            const _Float16* bp = wrow + (size_t)k * D_DIM + kk + grp * 8;
            v8h blo = *(const v8h*)(bp);
            v8h bhi = *(const v8h*)(bp + 16);
            v16h bf = __builtin_shufflevector(blo, bhi,
                        0, 1, 2, 3, 4, 5, 6, 7, 8, 9, 10, 11, 12, 13, 14, 15);
            acc[t] = __builtin_amdgcn_wmma_f32_16x16x32_f16(
                         false, a, false, bf, (short)0, acc[t], false, false);
        }
    }

    // --- store D: VGPR r, lanes 0-15: M=r,N=lane ; lanes 16-31: M=r+8,N=lane-16
    #pragma unroll
    for (int t = 0; t < 4; ++t) {
        const int k  = kbase + t * 16 + m;   // 16-aligned tile: same ni for all n
        const int ni = k >> 7;
        const int d  = k & 127;
        #pragma unroll
        for (int r = 0; r < 8; ++r) {
            const int mrow = r + grp * 8;
            const int b    = btile * 16 + mrow;
            hatf[(((size_t)b * NCAP + ni) * S_LEN + s) * D_DIM + d] =
                (_Float16)acc[t][r];
        }
    }
}

// ---------------------------------------------------------------------------
// Kernel 3: dynamic routing, one block (256 threads) per batch.
//   hat[b] (4*50*128 f16 = 51.2 KB) staged to LDS once; 3 routing iterations.
// ---------------------------------------------------------------------------
__global__ __launch_bounds__(256) void routing_kernel(
    const _Float16* __restrict__ hatf,
    const float*    __restrict__ hist_mask,
    float*          __restrict__ out)
{
    __shared__ _Float16 hat_s[NCAP * S_LEN * D_DIM];  // 51200 B
    __shared__ float w_s[NCAP * S_LEN];
    __shared__ float sw_s[NCAP * 52];
    __shared__ float mask_s[S_LEN];
    __shared__ float cap_s[NCAP * D_DIM];
    __shared__ float cn_s[NCAP];
    __shared__ float inter_s[NCAP * D_DIM];

    const int b   = blockIdx.x;
    const int tid = threadIdx.x;

    // stage hat[b] : contiguous 25600 halves = 3200 uint4
    {
        const uint4* src = (const uint4*)(hatf + (size_t)b * NCAP * S_LEN * D_DIM);
        uint4* dst = (uint4*)hat_s;
        for (int i = tid; i < 3200; i += 256) dst[i] = src[i];
    }
    if (tid < S_LEN) mask_s[tid] = hist_mask[b * S_LEN + tid];
    if (tid < NCAP * S_LEN) w_s[tid] = 0.0f;
    __syncthreads();

    for (int iter = 0; iter < 3; ++iter) {
        // softmax over n (axis=1) per s, then zero by mask
        if (tid < S_LEN) {
            float w0 = w_s[0 * S_LEN + tid], w1 = w_s[1 * S_LEN + tid];
            float w2 = w_s[2 * S_LEN + tid], w3 = w_s[3 * S_LEN + tid];
            float mx = fmaxf(fmaxf(w0, w1), fmaxf(w2, w3));
            float e0 = __expf(w0 - mx), e1 = __expf(w1 - mx);
            float e2 = __expf(w2 - mx), e3 = __expf(w3 - mx);
            float inv = 1.0f / (e0 + e1 + e2 + e3);
            float mk = (mask_s[tid] == 0.0f) ? 0.0f : 1.0f;
            sw_s[0 * 52 + tid] = e0 * inv * mk;
            sw_s[1 * 52 + tid] = e1 * inv * mk;
            sw_s[2 * 52 + tid] = e2 * inv * mk;
            sw_s[3 * 52 + tid] = e3 * inv * mk;
        }
        __syncthreads();

        // cap[n,d] = sum_s sw[n,s] * hat[n,s,d]
        for (int o = tid; o < NCAP * D_DIM; o += 256) {
            const int n = o >> 7, d = o & 127;
            float acc = 0.0f;
            for (int ss = 0; ss < S_LEN; ++ss)
                acc += sw_s[n * 52 + ss] * (float)hat_s[(n * S_LEN + ss) * D_DIM + d];
            cap_s[o] = acc;
        }
        __syncthreads();

        // cn[n] = sum_d cap^2
        if (tid < NCAP) {
            float cn = 0.0f;
            for (int d = 0; d < D_DIM; ++d) {
                float c = cap_s[tid * D_DIM + d];
                cn += c * c;
            }
            cn_s[tid] = cn;
        }
        __syncthreads();

        // squash
        for (int o = tid; o < NCAP * D_DIM; o += 256) {
            const float cn = cn_s[o >> 7];
            const float scale = cn / (1.0f + cn) / sqrtf(cn + 1e-9f);
            inter_s[o] = cap_s[o] * scale;
        }
        __syncthreads();

        if (iter < 2) {
            // delta[n,s] = sum_d hat[n,s,d] * inter[n,d] ; w += delta
            if (tid < NCAP * S_LEN) {
                const int n = tid / S_LEN, ss = tid % S_LEN;
                float acc = 0.0f;
                for (int d = 0; d < D_DIM; ++d)
                    acc += (float)hat_s[(n * S_LEN + ss) * D_DIM + d]
                           * inter_s[n * D_DIM + d];
                w_s[tid] += acc;
            }
            __syncthreads();
        }
    }

    // write interest_capsule (B, NI, D) f32
    for (int o = tid; o < NCAP * D_DIM; o += 256)
        out[(size_t)b * NCAP * D_DIM + o] = inter_s[o];
}

// ---------------------------------------------------------------------------
extern "C" void kernel_launch(void* const* d_in, const int* in_sizes, int n_in,
                              void* d_out, int out_size, void* d_ws, size_t ws_size,
                              hipStream_t stream) {
    // inputs: 0 nick_id, 1 item_id, 2 hist_item, 3 hist_mask, 4 user_emb,
    //         5 mid_emb, 6 caps_w   (0,1,4 unused by the reference)
    const int*   hist_item = (const int*)  d_in[2];
    const float* hist_mask = (const float*)d_in[3];
    const float* mid_emb   = (const float*)d_in[5];
    const float* caps_w    = (const float*)d_in[6];
    float* out = (float*)d_out;

    const int n_w = S_LEN * KTOT * D_DIM;                 // 3,276,800
    _Float16* wf16 = (_Float16*)d_ws;
    _Float16* hatf = (_Float16*)((char*)d_ws + (size_t)n_w * 2);  // 6,553,600 B offset

    convert_w_kernel<<<(n_w / 4 + 255) / 256, 256, 0, stream>>>(caps_w, wf16, n_w);

    dim3 g2(B_TOT / 16, S_LEN);
    hat_wmma_kernel<<<g2, 256, 0, stream>>>(hist_item, hist_mask, mid_emb, wf16, hatf);

    routing_kernel<<<B_TOT, 256, 0, stream>>>(hatf, hist_mask, out);
}
